// GCNModel_12584254177938
// MI455X (gfx1250) — matrix operand order repaired and checked by reading, not compiled, not run on hardware
//
#include <hip/hip_runtime.h>
#include <stdint.h>

typedef __attribute__((ext_vector_type(16))) _Float16 v16h;
typedef __attribute__((ext_vector_type(8)))  _Float16 v8h;
typedef __attribute__((ext_vector_type(8)))  float    v8f;

#define INDIM 512
#define HID0  96
#define HID2  384
#define OUTD  40
#define OUTP  48   // OUTD padded to 16

__device__ __forceinline__ void atomAddF32(float* p, float v) {
    // lowers to GLOBAL_ATOMIC_ADD_F32 (no-return form) on gfx1250
    __hip_atomic_fetch_add(p, v, __ATOMIC_RELAXED, __HIP_MEMORY_SCOPE_AGENT);
}

// ---------------- degree pipeline ----------------
__global__ void k_fill1(float* deg, int n) {
    int i = blockIdx.x * blockDim.x + threadIdx.x;
    if (i < n) deg[i] = 1.0f;   // +1 self loop
}
__global__ void k_deg(const long long* __restrict__ ei, float* deg, long long E) {
    long long e = (long long)blockIdx.x * blockDim.x + threadIdx.x;
    if (e < E) atomAddF32(&deg[ei[E + e]], 1.0f);
}
__global__ void k_rsqrt(const float* __restrict__ deg, float* dinv, int n) {
    int i = blockIdx.x * blockDim.x + threadIdx.x;
    if (i < n) dinv[i] = rsqrtf(deg[i]);
}

// ---------------- weight pre-pack ----------------
// Bp[kt][n][klocal] : f16, 32 contiguous K per column, N zero-padded to Npad.
// grid.x = K, blockDim.x = Npad
__global__ void k_packB(const float* __restrict__ B, _Float16* __restrict__ Bp,
                        int Nd, int Npad) {
    int k = blockIdx.x, n = threadIdx.x;
    float v = (n < Nd) ? B[(size_t)k * Nd + n] : 0.0f;
    Bp[(((size_t)(k >> 5) * Npad + n) << 5) + (k & 31)] = (_Float16)v;
}

// ---------------- WMMA GEMM: C[M,Nd] = A[M,K] @ B[K,Nd] (+epilogue) --------
// One wave computes one 16x16 tile. K multiple of 32, M multiple of 16,
// B pre-packed f16 with Npad columns.
// Inner loop (AF16): 4 vmem b128 + 1 WMMA. (f32 A: 6 b128 + cvt + 1 WMMA)
// EPI: 0 = raw f32 store; 1 = bias+relu -> f16 store; 2 = bias -> f32 store
template<bool AF16, int EPI>
__global__ void k_gemm(const void* __restrict__ Aptr,
                       const _Float16* __restrict__ Bp,
                       const float* __restrict__ bias,
                       void* __restrict__ Outp,
                       int M, int K, int Nd, int Npad, int ntn) {
    const int wave = threadIdx.x >> 5;
    const int lane = threadIdx.x & 31;
    int tile = blockIdx.x * (blockDim.x >> 5) + wave;
    int total = (M >> 4) * ntn;
    if (tile >= total) return;
    const int mt  = tile / ntn;
    const int nt  = tile % ntn;
    const int hi  = lane >> 4;          // 0 or 1 (lane half)
    const int l15 = lane & 15;
    const int m    = (mt << 4) + l15;   // A row this lane feeds
    const int ncol = (nt << 4) + l15;   // B column this lane feeds (< Npad always)
    const int aoff = hi << 3;           // A K-offset: 0 / 8
    const int boff = hi << 3;           // B K-offset: 0 / 8

    v8f acc = {};
    for (int kk = 0; kk < K; kk += 32) {
        v16h av, bv;
        // ---- A fragment (16-bit A 16x32 layout: i<8 -> K=aoff+i, i>=8 -> K=aoff+8+i)
        if constexpr (!AF16) {
            const float* ap = (const float*)Aptr + (size_t)m * K + kk + aoff;
            float4 p0 = *(const float4*)(ap);
            float4 p1 = *(const float4*)(ap + 4);
            float4 p2 = *(const float4*)(ap + 16);
            float4 p3 = *(const float4*)(ap + 20);
            av[0]=(_Float16)p0.x;  av[1]=(_Float16)p0.y;  av[2]=(_Float16)p0.z;  av[3]=(_Float16)p0.w;
            av[4]=(_Float16)p1.x;  av[5]=(_Float16)p1.y;  av[6]=(_Float16)p1.z;  av[7]=(_Float16)p1.w;
            av[8]=(_Float16)p2.x;  av[9]=(_Float16)p2.y;  av[10]=(_Float16)p2.z; av[11]=(_Float16)p2.w;
            av[12]=(_Float16)p3.x; av[13]=(_Float16)p3.y; av[14]=(_Float16)p3.z; av[15]=(_Float16)p3.w;
        } else {
            const _Float16* ap = (const _Float16*)Aptr + (size_t)m * K + kk + aoff;
            v8h q0 = *(const v8h*)(ap);       // K = +0..7
            v8h q1 = *(const v8h*)(ap + 16);  // K = +16..23
#pragma unroll
            for (int i = 0; i < 8; ++i) { av[i] = q0[i]; av[8 + i] = q1[i]; }
        }
        // ---- B fragment: i<8 -> K=boff+i, i>=8 -> K=boff+8+i (col = ncol)
        const _Float16* bp = Bp + (((size_t)(kk >> 5) * Npad + ncol) << 5) + boff;
        v8h r0 = *(const v8h*)(bp);
        v8h r1 = *(const v8h*)(bp + 16);
#pragma unroll
        for (int i = 0; i < 8; ++i) { bv[i] = r0[i]; bv[8 + i] = r1[i]; }

        acc = __builtin_amdgcn_wmma_f32_16x16x32_f16(false, av, false, bv,
                                                     (short)0, acc, false, false);
    }

    if (ncol >= Nd) return;  // store guard only (EXEC full through all WMMAs)
    float bval = 0.0f;
    if constexpr (EPI != 0) bval = bias[ncol];
#pragma unroll
    for (int j = 0; j < 8; ++j) {
        // C/D layout: VGPR j holds M = j (lanes 0-15), M = j+8 (lanes 16-31)
        int mr = (mt << 4) + j + (hi << 3);
        float v = acc[j];
        if constexpr (EPI == 0) {
            ((float*)Outp)[(size_t)mr * Nd + ncol] = v;
        } else if constexpr (EPI == 1) {
            v += bval; v = v > 0.0f ? v : 0.0f;
            ((_Float16*)Outp)[(size_t)mr * Nd + ncol] = (_Float16)v;
        } else {
            ((float*)Outp)[(size_t)mr * Nd + ncol] = v + bval;
        }
    }
}

// ---------------- graph aggregation ----------------
// agg[i,:] = t[i,:] * dinv[i]^2   (self-loop term, also initializes agg)
__global__ void k_selfloop(const float* __restrict__ t, const float* __restrict__ dinv,
                           float* __restrict__ agg, int Nd) {
    int row = blockIdx.x, d = threadIdx.x;
    float di = dinv[row];
    agg[(size_t)row * Nd + d] = t[(size_t)row * Nd + d] * di * di;
}
// agg[dst,:] += t[src,:] * dinv[src]*dinv[dst]   (one 96-thread block per edge)
__global__ void k_edge_agg(const long long* __restrict__ ei, const float* __restrict__ dinv,
                           const float* __restrict__ t, float* __restrict__ agg,
                           long long E, int Nd) {
    long long e = blockIdx.x;
    int d = threadIdx.x;
    long long s  = ei[e];
    long long dn = ei[E + e];
    float coef = dinv[s] * dinv[dn];
    atomAddF32(&agg[(size_t)dn * Nd + d], t[(size_t)s * Nd + d] * coef);
}
// h_f16[i,:] = sigmoid(agg[i,:] + b)  -- f16 output feeds the next GEMM's A side
__global__ void k_bias_sigmoid_h(const float* __restrict__ agg, const float* __restrict__ b,
                                 _Float16* __restrict__ h, int Nd) {
    int row = blockIdx.x, d = threadIdx.x;
    size_t i = (size_t)row * Nd + d;
    float v = agg[i] + b[d];
    h[i] = (_Float16)(1.0f / (1.0f + __expf(-v)));
}

extern "C" void kernel_launch(void* const* d_in, const int* in_sizes, int n_in,
                              void* d_out, int out_size, void* d_ws, size_t ws_size,
                              hipStream_t stream) {
    const float*     X   = (const float*)d_in[0];
    const long long* ei  = (const long long*)d_in[1];   // [2, E] int64
    const float*     W1  = (const float*)d_in[2];
    const float*     b1  = (const float*)d_in[3];
    const float*     W2  = (const float*)d_in[4];
    const float*     b2  = (const float*)d_in[5];
    const float*     Wm1 = (const float*)d_in[6];
    const float*     bm1 = (const float*)d_in[7];
    const float*     Wm2 = (const float*)d_in[8];
    const float*     bm2 = (const float*)d_in[9];

    const int       Nn = in_sizes[0] / INDIM;   // 50000 (multiple of 16)
    const long long E  = in_sizes[1] / 2;       // 800000

    // workspace carve-up (~87 MB)
    float*    deg  = (float*)d_ws;
    float*    dinv = deg  + Nn;
    float*    bufA = dinv + Nn;                             // [Nn,96] f32  (GEMM out t)
    float*    bufB = bufA + (size_t)Nn * HID0;              // [Nn,96] f32  (aggregate)
    _Float16* bufH = (_Float16*)(bufB + (size_t)Nn * HID0); // [Nn,96] f16  (sigmoid out)
    _Float16* bufC = bufH + (size_t)Nn * HID0;              // [Nn,384] f16 (relu out)
    _Float16* W1p  = bufC + (size_t)Nn * HID2;              // [512*96]  f16 packed
    _Float16* W2p  = W1p + (size_t)INDIM * HID0;            // [96*96]
    _Float16* Wm1p = W2p + (size_t)HID0 * HID0;             // [96*384]
    _Float16* Wm2p = Wm1p + (size_t)HID0 * HID2;            // [384*48] (N padded)

    // pack weights to f16 K-tile-major (tiny, once per launch)
    k_packB<<<INDIM, HID0, 0, stream>>>(W1,  W1p,  HID0, HID0);
    k_packB<<<HID0,  HID0, 0, stream>>>(W2,  W2p,  HID0, HID0);
    k_packB<<<HID0,  HID2, 0, stream>>>(Wm1, Wm1p, HID2, HID2);
    k_packB<<<HID2,  OUTP, 0, stream>>>(Wm2, Wm2p, OUTD, OUTP);

    // degrees: deg = 1 + indegree; dinv = rsqrt(deg)
    k_fill1<<<(Nn + 255) / 256, 256, 0, stream>>>(deg, Nn);
    k_deg  <<<(int)((E + 255) / 256), 256, 0, stream>>>(ei, deg, E);
    k_rsqrt<<<(Nn + 255) / 256, 256, 0, stream>>>(deg, dinv, Nn);

    auto blocksFor = [](int tiles) { return (tiles + 7) / 8; };  // 8 waves / block

    // ---- GCN layer 1: t = X @ W1 ; aggregate ; h1 = sigmoid(agg + b1) -> f16
    {
        int ntn = HID0 / 16, tiles = (Nn / 16) * ntn;
        k_gemm<false, 0><<<blocksFor(tiles), 256, 0, stream>>>(
            X, W1p, nullptr, bufA, Nn, INDIM, HID0, HID0, ntn);
    }
    k_selfloop<<<Nn, HID0, 0, stream>>>(bufA, dinv, bufB, HID0);
    k_edge_agg<<<(int)E, HID0, 0, stream>>>(ei, dinv, bufA, bufB, E, HID0);
    k_bias_sigmoid_h<<<Nn, HID0, 0, stream>>>(bufB, b1, bufH, HID0);

    // ---- GCN layer 2: t = h1 @ W2 (f16 A) ; aggregate ; h2 = sigmoid -> f16
    {
        int ntn = HID0 / 16, tiles = (Nn / 16) * ntn;
        k_gemm<true, 0><<<blocksFor(tiles), 256, 0, stream>>>(
            bufH, W2p, nullptr, bufA, Nn, HID0, HID0, HID0, ntn);
    }
    k_selfloop<<<Nn, HID0, 0, stream>>>(bufA, dinv, bufB, HID0);
    k_edge_agg<<<(int)E, HID0, 0, stream>>>(ei, dinv, bufA, bufB, E, HID0);
    k_bias_sigmoid_h<<<Nn, HID0, 0, stream>>>(bufB, b2, bufH, HID0);

    // ---- MLP head: relu(h2 @ Wm1 + bm1) stored f16, then @ Wm2 + bm2 -> f32 out
    {
        int ntn = HID2 / 16, tiles = (Nn / 16) * ntn;
        k_gemm<true, 1><<<blocksFor(tiles), 256, 0, stream>>>(
            bufH, Wm1p, bm1, bufC, Nn, HID0, HID2, HID2, ntn);
    }
    {
        int ntn = OUTP / 16, tiles = (Nn / 16) * ntn;   // N=40 handled by pad + store guard
        k_gemm<true, 2><<<blocksFor(tiles), 256, 0, stream>>>(
            bufC, Wm2p, bm2, d_out, Nn, HID2, OUTD, OUTP, ntn);
    }
}
